// AttentionLayer_48026324304219
// MI455X (gfx1250) — matrix operand order repaired and checked by reading, not compiled, not run on hardware
//
#include <hip/hip_runtime.h>

// ---- problem constants ----
#define L_SEQ 4096
#define DIM   512
#define HID   512
#define PAD   16
#define NWIN  32          // 2P window positions (center excluded)
#define NCOL  1024        // concat(Ww, Wu) rows
#define UCOLS (L_SEQ + 2*PAD)   // 4128 padded columns of U^T

typedef __attribute__((ext_vector_type(16))) _Float16 v16h;
typedef __attribute__((ext_vector_type(8)))  float    v8f;

union F16Frag { v16h v; uint4 u[2]; };

// ---------------- conversion: f32 -> f16 ----------------
__global__ void cvt_f16_kernel(const float* __restrict__ src,
                               _Float16* __restrict__ dst, int n) {
    int i = blockIdx.x * blockDim.x + threadIdx.x;
    if (i < n) dst[i] = (_Float16)src[i];
}

// ---------------- zero the +/-P pad columns of U^T ----------------
__global__ void zero_pad_kernel(float* __restrict__ UXt) {
    int t = blockIdx.x * blockDim.x + threadIdx.x;   // 512*32 threads
    int h = t >> 5;
    int j = t & 31;
    int c = (j < PAD) ? j : (L_SEQ + j);             // cols [0,16) and [4112,4128)
    UXt[h * UCOLS + c] = 0.0f;
}

// ---------------- WMMA GEMM: C[L x 1024] = Xh * Wh^T ----------------
// n <  512 : WX[l][n]            (row-major)
// n >= 512 : UXt[n-512][P + l]   (transposed, padded columns)
__global__ __launch_bounds__(128)
void gemm_wmma_kernel(const _Float16* __restrict__ xh,   // [L][512]
                      const _Float16* __restrict__ wh,   // [1024][512]
                      float* __restrict__ WX,            // [L][512]
                      float* __restrict__ UXt)           // [512][4128]
{
    const int lane = threadIdx.x & 31;
    const int wave = threadIdx.x >> 5;
    const int tile = blockIdx.x * 4 + wave;   // 256*64 = 16384 tiles
    const int nt = tile & 63;                 // N/16
    const int mt = tile >> 6;                 // L/16
    const int lg = lane >> 4;                 // lane group (0/1)
    const int ln = lane & 15;

    const _Float16* aptr = xh + (size_t)(mt * 16 + ln) * DIM;  // A row (M = ln)
    const _Float16* bptr = wh + (size_t)(nt * 16 + ln) * DIM;  // B col (N = ln)

    v8f acc = {};
    #pragma unroll 4
    for (int kt = 0; kt < DIM; kt += 32) {
        // A 16x32 f16 layout: lane-group selects K+8; elems 0-7 <- K+0..7, 8-15 <- K+16..23
        F16Frag a;
        const int ka = kt + 8 * lg;
        a.u[0] = *(const uint4*)(aptr + ka);
        a.u[1] = *(const uint4*)(aptr + ka + 16);
        // B 32x16 f16 layout: lane-group selects K+8; elems 0-7 <- K+0..7, 8-15 <- K+16..23
        F16Frag b;
        const int kb = kt + 8 * lg;
        b.u[0] = *(const uint4*)(bptr + kb);
        b.u[1] = *(const uint4*)(bptr + kb + 16);
        acc = __builtin_amdgcn_wmma_f32_16x16x32_f16(
            /*neg_a=*/false, a.v, /*neg_b=*/false, b.v,
            /*c_mod=*/(short)0, acc, /*reuse_a=*/false, /*reuse_b=*/false);
    }

    // C/D layout: N = lane%16, M = r + 8*(lane/16)
    const int ng    = nt * 16 + ln;
    const int mbase = mt * 16 + 8 * lg;
    if (ng < HID) {
        #pragma unroll
        for (int r = 0; r < 8; ++r)
            WX[(size_t)(mbase + r) * HID + ng] = acc[r];
    } else {
        float* dst = UXt + (size_t)(ng - HID) * UCOLS + PAD + mbase;
        float4 s0 = make_float4(acc[0], acc[1], acc[2], acc[3]);
        float4 s1 = make_float4(acc[4], acc[5], acc[6], acc[7]);
        *(float4*)(dst)     = s0;
        *(float4*)(dst + 4) = s1;
    }
}

// ---------------- window scores + softmax + weighted sum ----------------
// One wave per sequence row l.  Phase 1: lane = window position w.
// Phase 2: lane = feature chunk.
__global__ __launch_bounds__(256)
void attn_window_kernel(const float* __restrict__ WX,    // [4096][512]
                        const float* __restrict__ UXt,   // [512][4128]
                        const float* __restrict__ Wv,    // [512]
                        const float* __restrict__ x,     // [4096][512]
                        float* __restrict__ g)           // [4096][512]
{
    const int lane = threadIdx.x & 31;
    const int wave = threadIdx.x >> 5;
    const int l    = blockIdx.x * 8 + wave;

    // window offset for this lane: {0..15, 17..32} (center excluded)
    const int off = (lane < PAD) ? lane : lane + 1;
    const int col = l + off;                    // padded U^T column, in [0, 4127]

    const float* wxrow = WX + (size_t)l * HID;

    // score_w = sum_h Wv[h] * tanh(wx[l][h] + U^T[h][l+off_w])
    float score = 0.0f;
    #pragma unroll 4
    for (int h = 0; h < HID; ++h) {
        float u = UXt[(size_t)h * UCOLS + col];   // coalesced 33-float span
        score += Wv[h] * tanhf(wxrow[h] + u);
    }

    // stable softmax across the 32 lanes (wave32 shuffle reduction)
    float m = score;
    #pragma unroll
    for (int d = 16; d > 0; d >>= 1) m = fmaxf(m, __shfl_xor(m, d, 32));
    float e = __expf(score - m);
    float s = e;
    #pragma unroll
    for (int d = 16; d > 0; d >>= 1) s += __shfl_xor(s, d, 32);
    const float attn = e / s;

    // g[l][:] = sum_w attn_w * x[l + off_w - P][:]   (lane = feature chunk)
    float4 acc0 = make_float4(0.f, 0.f, 0.f, 0.f);
    float4 acc1 = acc0, acc2 = acc0, acc3 = acc0;
    #pragma unroll
    for (int w = 0; w < NWIN; ++w) {
        const float aw = __shfl(attn, w, 32);
        const int row = l + ((w < PAD) ? w : w + 1) - PAD;
        if (row >= 0 && row < L_SEQ) {
            const float4* xr = (const float4*)(x + (size_t)row * DIM);
            float4 v0 = xr[lane];
            float4 v1 = xr[lane + 32];
            float4 v2 = xr[lane + 64];
            float4 v3 = xr[lane + 96];
            acc0.x += aw * v0.x; acc0.y += aw * v0.y; acc0.z += aw * v0.z; acc0.w += aw * v0.w;
            acc1.x += aw * v1.x; acc1.y += aw * v1.y; acc1.z += aw * v1.z; acc1.w += aw * v1.w;
            acc2.x += aw * v2.x; acc2.y += aw * v2.y; acc2.z += aw * v2.z; acc2.w += aw * v2.w;
            acc3.x += aw * v3.x; acc3.y += aw * v3.y; acc3.z += aw * v3.z; acc3.w += aw * v3.w;
        }
    }
    float4* gr = (float4*)(g + (size_t)l * DIM);
    gr[lane]      = acc0;
    gr[lane + 32] = acc1;
    gr[lane + 64] = acc2;
    gr[lane + 96] = acc3;
}

// ---------------- launch ----------------
extern "C" void kernel_launch(void* const* d_in, const int* in_sizes, int n_in,
                              void* d_out, int out_size, void* d_ws, size_t ws_size,
                              hipStream_t stream) {
    const float* x  = (const float*)d_in[0];   // [4096*512]
    const float* Ww = (const float*)d_in[1];   // [512*512]
    const float* Wu = (const float*)d_in[2];   // [512*512]
    const float* Wv = (const float*)d_in[3];   // [512]
    float* g = (float*)d_out;

    char* ws = (char*)d_ws;
    // workspace layout (all offsets 256B aligned)
    _Float16* xh  = (_Float16*)(ws);                       // 4096*512*2  = 4 MiB
    _Float16* wh  = (_Float16*)(ws + 4194304);             // 1024*512*2  = 1 MiB
    float*    WX  = (float*)   (ws + 5242880);             // 4096*512*4  = 8 MiB
    float*    UXt = (float*)   (ws + 13631488);            // 512*4128*4  ~ 8.06 MiB

    const int nX = L_SEQ * DIM;   // 2,097,152
    const int nW = HID * DIM;     //   262,144

    cvt_f16_kernel<<<(nX + 255) / 256, 256, 0, stream>>>(x,  xh, nX);
    cvt_f16_kernel<<<(nW + 255) / 256, 256, 0, stream>>>(Ww, wh, nW);
    cvt_f16_kernel<<<(nW + 255) / 256, 256, 0, stream>>>(Wu, wh + nW, nW);
    zero_pad_kernel<<<(HID * 32) / 256, 256, 0, stream>>>(UXt);

    // 16384 16x16 tiles, 4 waves (tiles) per 128-thread block
    gemm_wmma_kernel<<<4096, 128, 0, stream>>>(xh, wh, WX, UXt);

    // one wave per row, 8 waves per block
    attn_window_kernel<<<L_SEQ / 8, 256, 0, stream>>>(WX, UXt, Wv, x, g);
}
